// GNN_34239479283984
// MI455X (gfx1250) — matrix operand order repaired and checked by reading, not compiled, not run on hardware
//
#include <hip/hip_runtime.h>
#include <hip/hip_bf16.h>

// ---------------------------------------------------------------------------
// Fused GNN node pipeline for MI455X (gfx1250, wave32, WMMA f32_16x16x32_f16)
//
// ~35.6 GFLOP vs ~36 MB HBM traffic -> strongly compute bound. All 4 layers
// fused; the four [N x 64]x[64 x 64] GEMMs run on v_wmma_f32_16x16x32_f16.
// Weights live in LDS pre-swizzled into WMMA fragment order so every
// B-fragment is one contiguous 32B v16h load per lane (2x ds_load_b128,
// bank-conflict-free). Activations use a 72-half stride so A-fragments are
// 16B-aligned v8h loads. C operand uses the inline-constant 0 (keeps VGPR
// pressure inside the 256-direct window; bias+relu dual-issues in epilogue).
// Pooling is deterministic (no atomics).
// ---------------------------------------------------------------------------

typedef __attribute__((ext_vector_type(16))) _Float16 v16h;
typedef __attribute__((ext_vector_type(8)))  _Float16 v8h;
typedef __attribute__((ext_vector_type(8)))  float    v8f;

#define ACT_S 72   // activation row stride in halves: 16B-aligned chunks, 36-bank stride

__device__ __forceinline__ v8f wmma16(v16h a, v16h b, v8f c) {
    // (neg_a, A, neg_b, B, c_mod, C, reuse_a, reuse_b)
    return __builtin_amdgcn_wmma_f32_16x16x32_f16(false, a, false, b,
                                                  (short)0, c, false, false);
}

// A fragment (16x32 f16). ISA layout: lanes 0-15 hold M=lane; v16h elems
// 0..7 = K kbase+koff+0..7, elems 8..15 = K kbase+16+koff+0..7 (koff=8 hi half).
__device__ __forceinline__ v16h load_a_frag(const _Float16* act, int m, int kbase, int hi) {
    const _Float16* p = act + m * ACT_S + kbase + hi * 8;
    v8h lo = *(const v8h*)p;
    v8h hh = *(const v8h*)(p + 16);
    return __builtin_shufflevector(lo, hh, 0, 1, 2, 3, 4, 5, 6, 7,
                                   8, 9, 10, 11, 12, 13, 14, 15);
}

// B fragment from fragment-order storage: [..][lane][16 halves] contiguous.
__device__ __forceinline__ v16h load_fragB(const _Float16* base, int l, int nt, int ks, int lane) {
    const _Float16* p = base + ((((l * 4 + nt) * 2 + ks) * 32) + lane) * 16;
    return *(const v16h*)p;
}
__device__ __forceinline__ v16h load_fragA0(const _Float16* base, int nt, int lane) {
    const _Float16* p = base + (nt * 32 + lane) * 16;
    return *(const v16h*)p;
}

// Second GEMM of an MLP3 ([M x 64] x [64 x 64]) fused with bias+relu and the
// 64->1 projection (Wc), reduced across the 16-lane half via shfl_xor.
__device__ __forceinline__ void gemmB_proj(const _Float16* act, const _Float16* wfrag,
                                           int l, const float* bB, const float* wC,
                                           float cBias, bool residual, float* cbuf,
                                           int mBase, int ln, int hi, int lane) {
    const int mA = mBase + ln;
    v16h a0 = load_a_frag(act, mA, 0,  hi);
    v16h a1 = load_a_frag(act, mA, 32, hi);
    float pv[8] = {0.f, 0.f, 0.f, 0.f, 0.f, 0.f, 0.f, 0.f};
#pragma unroll
    for (int nt = 0; nt < 4; ++nt) {
        v16h b0 = load_fragB(wfrag, l, nt, 0, lane);
        v16h b1 = load_fragB(wfrag, l, nt, 1, lane);
        v8f acc = {};
        acc = wmma16(a0, b0, acc);
        acc = wmma16(a1, b1, acc);
        const int n = nt * 16 + ln;
        const float bias = bB[n];
        const float wc   = wC[n];
#pragma unroll
        for (int v = 0; v < 8; ++v)
            pv[v] += fmaxf(acc[v] + bias, 0.f) * wc;   // relu(h2) * Wc
    }
#pragma unroll
    for (int s = 1; s < 16; s <<= 1) {
#pragma unroll
        for (int v = 0; v < 8; ++v) pv[v] += __shfl_xor(pv[v], s, 32);
    }
    if (ln == 0) {
#pragma unroll
        for (int v = 0; v < 8; ++v) {
            const int m = mBase + hi * 8 + v;
            float val = pv[v] + cBias;
            if (residual) val += cbuf[m];
            cbuf[m] = val;
        }
    }
}

extern "C" __global__ void __launch_bounds__(256)
gnn_node_kernel(const float* __restrict__ x, const float* __restrict__ u,
                const int* __restrict__ batch,
                const float* __restrict__ w0a, const float* __restrict__ b0a,
                const float* __restrict__ w0b, const float* __restrict__ b0b,
                const float* __restrict__ w0c, const float* __restrict__ b0c,
                const float* __restrict__ Wa,  const float* __restrict__ Ba,
                const float* __restrict__ Wb,  const float* __restrict__ Bb,
                const float* __restrict__ Wc,  const float* __restrict__ Bc,
                float* __restrict__ gPart, int N, int NT) {
    // weights pre-swizzled to fragment order: one contiguous 32B chunk per lane
    __shared__ __attribute__((aligned(32))) _Float16 sWA0f[4 * 32 * 16];        // 4 KB
    __shared__ __attribute__((aligned(32))) _Float16 sWBf[4 * 4 * 2 * 32 * 16]; // 32 KB
    __shared__ __attribute__((aligned(32))) _Float16 act[128 * ACT_S];          // 18 KB
    __shared__ float sb0a[64];
    __shared__ float sBB[4 * 64];             // b0b, Bb[0..2]
    __shared__ float sWa[576];                // Wa [3][3][64] f32 (VALU path)
    __shared__ float sBa[192];                // Ba [3][64]
    __shared__ float sWC[4 * 64];             // w0c, Wc[0..2]
    __shared__ float sBc[4];                  // b0c, Bc[0..2]
    __shared__ float cbuf[128];               // per-node carry scalar
    __shared__ float ugbuf[256];              // per-node u gather [m][2]
    __shared__ int   bbuf[128];               // per-node graph id (-1 = invalid)
    __shared__ float pool[256];               // block-local segment sums
    __shared__ float red[4];                  // pooling wave partials

    const int t = threadIdx.x;

    // ---- one-time weight staging (f32 -> f16, fragment-order swizzle) ----
    // sWBf index: [l(2b)][nt(2b)][ks(1b)][lane(5b)][j(4b)]
    for (int i = t; i < 4 * 4 * 2 * 32 * 16; i += 256) {
        const int j    = i & 15;
        const int lane = (i >> 4) & 31;
        const int ks   = (i >> 9) & 1;
        const int nt   = (i >> 10) & 3;
        const int l    = i >> 12;
        const int n = nt * 16 + (lane & 15);
        const int k = ks * 32 + (j >> 3) * 16 + (lane >> 4) * 8 + (j & 7);
        const float* src = (l == 0) ? w0b : (Wb + (l - 1) * 4096);
        sWBf[i] = (_Float16)src[k * 64 + n];
    }
    // sWA0f index: [nt(2b)][lane(5b)][j(4b)], K padded 9 -> 32
    for (int i = t; i < 4 * 32 * 16; i += 256) {
        const int j    = i & 15;
        const int lane = (i >> 4) & 31;
        const int nt   = i >> 9;
        const int n = nt * 16 + (lane & 15);
        const int k = (j >> 3) * 16 + (lane >> 4) * 8 + (j & 7);
        sWA0f[i] = (k < 9) ? (_Float16)w0a[k * 64 + n] : (_Float16)0.f;
    }
    if (t < 64) sb0a[t] = b0a[t];
    {
        const int j = t >> 6, n = t & 63;
        sBB[t] = (j == 0) ? b0b[n] : Bb[(j - 1) * 64 + n];
        sWC[t] = (j == 0) ? w0c[n] : Wc[(j - 1) * 64 + n];
    }
    for (int i = t; i < 576; i += 256) sWa[i] = Wa[i];
    if (t < 192) sBa[t] = Ba[t];
    if (t < 4)   sBc[t] = (t == 0) ? b0c[0] : Bc[t - 1];
    pool[t] = 0.f;
    __syncthreads();

    const int wave = t >> 5, lane = t & 31;
    const int ln = lane & 15, hi = lane >> 4;
    const int mBase = wave * 16;              // 8 waves * 16 rows = 128-node tile

    for (int tile = blockIdx.x; tile < NT; tile += gridDim.x) {
        // ---- stage inputs: act[m][0..31] = [x(7), ug(2), zeros] ----
        if (t < 128) {
            const int node = tile * 128 + t;
            float g0 = 0.f, g1 = 0.f;
            int b = -1;
            v8h p0 = {}, p1 = {}, pz = {};
            if (node < N) {
                b = batch[node];
                g0 = u[2 * b];
                g1 = u[2 * b + 1];
                const float* xr = x + node * 7;
#pragma unroll
                for (int j = 0; j < 7; ++j) p0[j] = (_Float16)xr[j];
                p0[7] = (_Float16)g0;
                p1[0] = (_Float16)g1;
            }
            v8h* row = (v8h*)(act + t * ACT_S);
            row[0] = p0; row[1] = p1; row[2] = pz; row[3] = pz;
            bbuf[t] = b;
            ugbuf[2 * t] = g0;
            ugbuf[2 * t + 1] = g1;
        }
        __syncthreads();

        // ---- layer 0, GEMM A: [128 x 32pad] x [32 x 64] -> h1 (relu) ----
        {
            v16h a = load_a_frag(act, mBase + ln, 0, hi);
#pragma unroll
            for (int nt = 0; nt < 4; ++nt) {
                v16h b = load_fragA0(sWA0f, nt, lane);
                v8f acc = {};
                acc = wmma16(a, b, acc);
                const int n = nt * 16 + ln;
                const float bias = sb0a[n];
#pragma unroll
                for (int v = 0; v < 8; ++v)
                    act[(mBase + hi * 8 + v) * ACT_S + n] =
                        (_Float16)fmaxf(acc[v] + bias, 0.f);
            }
        }
        // ---- layer 0, GEMM B + 64->1 projection -> cbuf ----
        gemmB_proj(act, sWBf, 0, sBB, sWC, sBc[0], false, cbuf, mBase, ln, hi, lane);

        // ---- residual layers: K=3 input layer on VALU, 64x64 on WMMA ----
#pragma unroll
        for (int l = 0; l < 3; ++l) {
            __syncthreads();
            {
                const int m = t >> 1, nb0 = (t & 1) * 32;
                const float cm = cbuf[m];
                const float g0 = ugbuf[2 * m], g1 = ugbuf[2 * m + 1];
                const float* wa = sWa + l * 192;     // [3][64]
                const float* ba = sBa + l * 64;
                v8h* row = (v8h*)(act + m * ACT_S + nb0);
#pragma unroll
                for (int c8 = 0; c8 < 4; ++c8) {
                    v8h pk;
#pragma unroll
                    for (int i = 0; i < 8; ++i) {
                        const int n = nb0 + c8 * 8 + i;
                        const float v =
                            cm * wa[n] + g0 * wa[64 + n] + g1 * wa[128 + n] + ba[n];
                        pk[i] = (_Float16)fmaxf(v, 0.f);
                    }
                    row[c8] = pk;
                }
            }
            __syncthreads();
            gemmB_proj(act, sWBf, l + 1, sBB + (l + 1) * 64, sWC + (l + 1) * 64,
                       sBc[l + 1], true, cbuf, mBase, ln, hi, lane);
        }

        __syncthreads();
        // ---- deterministic pooling into block-local pool ----
        const bool uni = (bbuf[0] == bbuf[127]) && (bbuf[0] >= 0);  // batch sorted
        if (uni) {
            float c = (t < 128) ? cbuf[t] : 0.f;
#pragma unroll
            for (int s = 1; s < 32; s <<= 1) c += __shfl_xor(c, s, 32);
            if (t < 128 && (t & 31) == 0) red[t >> 5] = c;
            __syncthreads();
            if (t == 0) pool[bbuf[0]] += (red[0] + red[1]) + (red[2] + red[3]);
        } else {
            if (t == 0) {
                for (int i = 0; i < 128; ++i) {
                    const int b = bbuf[i];
                    if (b >= 0) pool[b] += cbuf[i];
                }
            }
        }
        __syncthreads();
    }

    gPart[blockIdx.x * 256 + t] = pool[t];
}

// Fixed-order partial reduction + output MLP (256 graphs, one per thread).
extern "C" __global__ void __launch_bounds__(256)
gnn_finish_kernel(const float* __restrict__ part, int nb,
                  const float* __restrict__ Wo1, const float* __restrict__ bo1,
                  const float* __restrict__ Wo2, const float* __restrict__ bo2,
                  const float* __restrict__ Wo3, const float* __restrict__ bo3,
                  const float* __restrict__ Wo4, const float* __restrict__ bo4,
                  float* __restrict__ out) {
    __shared__ float sW[64 * 64];
    const int g = threadIdx.x;

    float p = 0.f;
    for (int k = 0; k < nb; ++k) p += part[k * 256 + g];  // deterministic order

    float h1[64], h2[64];
#pragma unroll
    for (int j = 0; j < 64; ++j) h1[j] = fmaxf(p * Wo1[j] + bo1[j], 0.f);

    for (int i = g; i < 4096; i += 256) sW[i] = Wo2[i];
    __syncthreads();
    for (int j = 0; j < 64; ++j) {
        float s = bo2[j];
#pragma unroll
        for (int i = 0; i < 64; ++i) s += h1[i] * sW[i * 64 + j];
        h2[j] = fmaxf(s, 0.f);
    }
    __syncthreads();
    for (int i = g; i < 4096; i += 256) sW[i] = Wo3[i];
    __syncthreads();
    for (int j = 0; j < 64; ++j) {
        float s = bo3[j];
#pragma unroll
        for (int i = 0; i < 64; ++i) s += h2[i] * sW[i * 64 + j];
        h1[j] = fmaxf(s, 0.f);
    }
    float o0 = bo4[0], o1 = bo4[1];
#pragma unroll
    for (int i = 0; i < 64; ++i) {
        o0 += h1[i] * Wo4[i * 2 + 0];
        o1 += h1[i] * Wo4[i * 2 + 1];
    }
    out[g * 2 + 0] = o0;
    out[g * 2 + 1] = o1;
}

extern "C" void kernel_launch(void* const* d_in, const int* in_sizes, int n_in,
                              void* d_out, int out_size, void* d_ws, size_t ws_size,
                              hipStream_t stream) {
    const float* x    = (const float*)d_in[0];
    const float* u    = (const float*)d_in[1];
    const int*  batch = (const int*)d_in[2];
    const float* w0a = (const float*)d_in[3];
    const float* b0a = (const float*)d_in[4];
    const float* w0b = (const float*)d_in[5];
    const float* b0b = (const float*)d_in[6];
    const float* w0c = (const float*)d_in[7];
    const float* b0c = (const float*)d_in[8];
    const float* Wa  = (const float*)d_in[9];
    const float* Ba  = (const float*)d_in[10];
    const float* Wb  = (const float*)d_in[11];
    const float* Bb  = (const float*)d_in[12];
    const float* Wc  = (const float*)d_in[13];
    const float* Bc  = (const float*)d_in[14];
    const float* Wo1 = (const float*)d_in[15];
    const float* bo1 = (const float*)d_in[16];
    const float* Wo2 = (const float*)d_in[17];
    const float* bo2 = (const float*)d_in[18];
    const float* Wo3 = (const float*)d_in[19];
    const float* bo3 = (const float*)d_in[20];
    const float* Wo4 = (const float*)d_in[21];
    const float* bo4 = (const float*)d_in[22];
    float* out = (float*)d_out;

    const int N  = in_sizes[0] / 7;
    const int NT = (N + 127) / 128;
    int NB = (int)(ws_size / (256 * sizeof(float)));
    if (NB > 1280) NB = 1280;
    if (NB > NT)   NB = NT;
    if (NB < 1)    NB = 1;
    float* gPart = (float*)d_ws;

    gnn_node_kernel<<<NB, 256, 0, stream>>>(x, u, batch, w0a, b0a, w0b, b0b,
                                            w0c, b0c, Wa, Ba, Wb, Bb, Wc, Bc,
                                            gPart, N, NT);
    gnn_finish_kernel<<<1, 256, 0, stream>>>(gPart, NB, Wo1, bo1, Wo2, bo2,
                                             Wo3, bo3, Wo4, bo4, out);
}
